// session_model_25091198943529
// MI455X (gfx1250) — matrix operand order repaired and checked by reading, not compiled, not run on hardware
//
#include <hip/hip_runtime.h>
#include <math.h>

typedef __attribute__((ext_vector_type(16))) __bf16 v16bf;
typedef __attribute__((ext_vector_type(8)))  float  v8f;

// ---------------------------------------------------------------------------
// Edge weight normalization: iw = w / in_deg[dst], ow = w / out_deg[src]
// One block per session (100 edges, 50 nodes, all local).
// ---------------------------------------------------------------------------
__global__ __launch_bounds__(128) void edge_norm_kernel(
    const int* __restrict__ src, const int* __restrict__ dst,
    const float* __restrict__ w, float* __restrict__ iw, float* __restrict__ ow) {
  __shared__ float indeg[50];
  __shared__ float outdeg[50];
  const int s = blockIdx.x;
  const int t = threadIdx.x;
  if (t < 50) { indeg[t] = 0.f; outdeg[t] = 0.f; }
  __syncthreads();
  const int ebase = s * 100;
  const int nbase = s * 50;
  if (t < 100) {
    const int e = ebase + t;
    const float we = w[e];
    atomicAdd(&indeg[dst[e] - nbase], we);
    atomicAdd(&outdeg[src[e] - nbase], we);
  }
  __syncthreads();
  if (t < 100) {
    const int e = ebase + t;
    const float we = w[e];
    iw[e] = we / indeg[dst[e] - nbase];
    ow[e] = we / outdeg[src[e] - nbase];
  }
}

// ---------------------------------------------------------------------------
// WMMA GEMM: C[M,NC] = A[M,K] @ W[K,NC] (+bias) (+=C if addC)
// Block = 256 threads (8 waves), 128 rows per block, wave owns 16 rows.
// W chunk (32 x NC) staged in LDS pre-swizzled to B-fragment layout
// (column-major, padded stride 40 bf16 -> one aligned 32B read per fragment).
// sched_group_barrier pins the schedule to: 8 ds_load_b128, then 4
// back-to-back v_wmma, per tile group -> one dscnt wait per 4 WMMAs and
// load/matrix overlap across groups.
// ---------------------------------------------------------------------------
template <int NC>
__global__ __launch_bounds__(256) void gemm_wmma_kernel(
    const float* __restrict__ A, const float* __restrict__ W,
    const float* __restrict__ bias, float* __restrict__ C,
    int K, int addC) {
  constexpr int LSTR = 40;                 // bf16 per column slot (padded from 32)
  __shared__ __bf16 wlds[NC * LSTR];
  const int tid  = threadIdx.x;
  const int wave = tid >> 5;
  const int lane = tid & 31;
  const int half = lane >> 4;              // 0: lanes 0-15, 1: lanes 16-31
  const int lm   = lane & 15;
  const long rowBase = (long)blockIdx.x * 128 + wave * 16;
  const float* Arow = A + (rowBase + lm) * (long)K;
  const __bf16* blds = &wlds[lm * LSTR + (half ? 16 : 0)];

  constexpr int NT = NC / 16;              // 16-col output tiles per wave
  v8f acc[NT];
#pragma unroll
  for (int t = 0; t < NT; ++t)
#pragma unroll
    for (int i = 0; i < 8; ++i) acc[t][i] = 0.f;

  for (int kb = 0; kb < K; kb += 32) {
    __syncthreads();
    // Stage W[kb..kb+31][:] -> wlds[n*LSTR + k] (global coalesced over n).
    for (int idx = tid; idx < 32 * NC; idx += 256) {
      const int k = idx / NC;
      const int n = idx % NC;              // NC power of 2 -> shifts
      wlds[n * LSTR + k] = (__bf16)W[(long)(kb + k) * NC + n];
    }
    __syncthreads();

    // A fragment: contiguous k order (same order as B below).
    // lanes 0-15: K = {0..15}; lanes 16-31: K = {16..31}
    const float4* ap = (const float4*)(Arow + kb + (half ? 16 : 0));
    const float4 a0 = ap[0];
    const float4 a1 = ap[1];
    const float4 a2 = ap[2];
    const float4 a3 = ap[3];
    v16bf afrag;
    afrag[0]  = (__bf16)a0.x; afrag[1]  = (__bf16)a0.y;
    afrag[2]  = (__bf16)a0.z; afrag[3]  = (__bf16)a0.w;
    afrag[4]  = (__bf16)a1.x; afrag[5]  = (__bf16)a1.y;
    afrag[6]  = (__bf16)a1.z; afrag[7]  = (__bf16)a1.w;
    afrag[8]  = (__bf16)a2.x; afrag[9]  = (__bf16)a2.y;
    afrag[10] = (__bf16)a2.z; afrag[11] = (__bf16)a2.w;
    afrag[12] = (__bf16)a3.x; afrag[13] = (__bf16)a3.y;
    afrag[14] = (__bf16)a3.z; afrag[15] = (__bf16)a3.w;

#pragma unroll
    for (int t0 = 0; t0 < NT; t0 += 4) {
      // B fragment: element e <-> K = half*16+e at column n = t*16+lm.
      v16bf bfr[4];
#pragma unroll
      for (int u = 0; u < 4; ++u)
        bfr[u] = *(const v16bf*)(blds + (t0 + u) * (16 * LSTR));
#pragma unroll
      for (int u = 0; u < 4; ++u)
        acc[t0 + u] = __builtin_amdgcn_wmma_f32_16x16x32_bf16(
            false, afrag, false, bfr[u], (short)0, acc[t0 + u], false, false);
      // Pin schedule: all 8 DS reads of this group first, then 4 WMMAs.
      __builtin_amdgcn_sched_group_barrier(0x100, 8, 0);  // DS read x8
      __builtin_amdgcn_sched_group_barrier(0x008, 4, 0);  // WMMA x4
    }
  }

  // C/D f32 layout: VGPR i -> row (i + half*8), col = lane%16
#pragma unroll
  for (int t = 0; t < NT; ++t) {
    const int n = t * 16 + lm;
    const float bv = bias ? bias[n] : 0.f;
#pragma unroll
    for (int i = 0; i < 8; ++i) {
      const long r   = rowBase + i + (half ? 8 : 0);
      const long off = r * NC + n;
      float v = acc[t][i] + bv;
      if (addC) v += C[off];
      C[off] = v;
    }
  }
}

// ---------------------------------------------------------------------------
// Per-session weighted scatter-add: a_in / a_out into a2 = [a_in | a_out]
// 128 threads = embedding dims; LDS accumulators (50 x 128) x2.
// ---------------------------------------------------------------------------
__global__ __launch_bounds__(128) void aggregate_kernel(
    const int* __restrict__ src, const int* __restrict__ dst,
    const float* __restrict__ iw, const float* __restrict__ ow,
    const float* __restrict__ fin, const float* __restrict__ fout,
    float* __restrict__ a2) {
  __shared__ float accIn[50 * 128];
  __shared__ float accOut[50 * 128];
  const int s = blockIdx.x;
  const int d = threadIdx.x;
  for (int i = 0; i < 50; ++i) { accIn[i * 128 + d] = 0.f; accOut[i * 128 + d] = 0.f; }
  const int ebase = s * 100;
  const int nbase = s * 50;
  for (int el = 0; el < 100; ++el) {
    const int e  = ebase + el;
    const int sn = src[e];
    const int dn = dst[e];
    accIn[(dn - nbase) * 128 + d]  += fin[(long)sn * 128 + d]  * iw[e];
    accOut[(sn - nbase) * 128 + d] += fout[(long)dn * 128 + d] * ow[e];
  }
  for (int i = 0; i < 50; ++i) {
    const long n = nbase + i;
    a2[n * 256 + d]       = accIn[i * 128 + d];
    a2[n * 256 + 128 + d] = accOut[i * 128 + d];
  }
}

// ---------------------------------------------------------------------------
// GRU-style gate + PReLU:  feat = prelu(ng + ig*(h - ng), a_gru)
// ---------------------------------------------------------------------------
__global__ __launch_bounds__(256) void gate_kernel(
    const float* __restrict__ f, const float* __restrict__ g2,
    const float* __restrict__ h0, const float* __restrict__ a_gru,
    float* __restrict__ feat, long total) {
  const long gid = (long)blockIdx.x * 256 + threadIdx.x;
  if (gid >= total) return;
  const long n = gid >> 7;
  const int  d = (int)(gid & 127);
  const float fi = f[n * 256 + d];
  const float fn = f[n * 256 + 128 + d];
  const float bz = g2[n * 256 + d];
  const float bh = g2[n * 256 + 128 + d];
  const float x  = h0[gid];
  const float ig = 1.f / (1.f + expf(-(fi + bz)));
  const float ng = tanhf(fn + bh);
  const float v  = ng + ig * (x - ng);
  feat[gid] = (v >= 0.f) ? v : a_gru[d] * v;
}

// ---------------------------------------------------------------------------
// Gather e_last rows; write left half of output.
// ---------------------------------------------------------------------------
__global__ __launch_bounds__(256) void gather_last_kernel(
    const float* __restrict__ feat, const int* __restrict__ last_nodes,
    float* __restrict__ elast, float* __restrict__ out) {
  const int gid = blockIdx.x * 256 + threadIdx.x;
  const int s = gid >> 7;
  const int d = gid & 127;
  const float v = feat[(long)last_nodes[s] * 128 + d];
  elast[gid] = v;
  out[(long)s * 256 + d] = v;
}

// ---------------------------------------------------------------------------
// PGAT attention readout per session: segment softmax over 50 nodes + PReLU.
// ---------------------------------------------------------------------------
__global__ __launch_bounds__(128) void attention_kernel(
    const float* __restrict__ qry, const float* __restrict__ kbuf,
    const float* __restrict__ feat, const float* __restrict__ W_e,
    const float* __restrict__ cnt, const float* __restrict__ a_att,
    float* __restrict__ out) {
  __shared__ float red[128];
  __shared__ float evals[50];
  const int s = blockIdx.x;
  const int d = threadIdx.x;
  const int nbase = s * 50;
  const float qv  = qry[(long)s * 128 + d];
  const float wev = W_e[d];
  for (int i = 0; i < 50; ++i) {
    const long n = nbase + i;
    const float t = wev / (1.f + expf(-(qv + kbuf[n * 128 + d])));
    red[d] = t;
    __syncthreads();
    for (int off = 64; off > 0; off >>= 1) {
      if (d < off) red[d] += red[d + off];
      __syncthreads();
    }
    if (d == 0) evals[i] = red[0] + logf(cnt[n]);
    __syncthreads();
  }
  float m = -1e30f;
  for (int i = 0; i < 50; ++i) m = fmaxf(m, evals[i]);
  float ssum = 0.f;
  for (int i = 0; i < 50; ++i) ssum += expf(evals[i] - m);
  float rst = 0.f;
  for (int i = 0; i < 50; ++i) {
    const float alpha = expf(evals[i] - m) / ssum;
    rst += alpha * feat[(long)(nbase + i) * 128 + d];
  }
  rst = (rst >= 0.f) ? rst : a_att[d] * rst;
  out[(long)s * 256 + 128 + d] = rst;
}

// ---------------------------------------------------------------------------
extern "C" void kernel_launch(void* const* d_in, const int* in_sizes, int n_in,
                              void* d_out, int out_size, void* d_ws, size_t ws_size,
                              hipStream_t stream) {
  const int B = 4096, P = 50, D = 128;
  const int N = B * P;      // 204800
  const int E = 2 * N;      // 409600
  (void)in_sizes; (void)n_in; (void)out_size; (void)ws_size; (void)P;

  const int*   src        = (const int*)d_in[0];
  const int*   dst        = (const int*)d_in[1];
  const int*   last_nodes = (const int*)d_in[2];
  /* graph_id d_in[3] unused: sessions are contiguous P-blocks */
  const float* w      = (const float*)d_in[4];
  const float* cnt    = (const float*)d_in[5];
  const float* h_i    = (const float*)d_in[6];
  const float* HKGU   = (const float*)d_in[7];
  const float* W_in   = (const float*)d_in[8];
  const float* b_in   = (const float*)d_in[9];
  const float* W_out  = (const float*)d_in[10];
  const float* b_out  = (const float*)d_in[11];
  const float* Wg_in  = (const float*)d_in[12];
  const float* bg_in  = (const float*)d_in[13];
  const float* Wg_out = (const float*)d_in[14];
  const float* a_gru  = (const float*)d_in[15];
  const float* W_user = (const float*)d_in[16];
  const float* b_user = (const float*)d_in[17];
  const float* W_key  = (const float*)d_in[18];
  const float* W_last = (const float*)d_in[19];
  const float* W_e    = (const float*)d_in[20];
  const float* a_att  = (const float*)d_in[21];
  float* out = (float*)d_out;

  // Workspace layout (floats), with lifetime-based reuse:
  float* ws  = (float*)d_ws;
  float* iw  = ws;                       // [E]
  float* ow  = iw + E;                   // [E]
  float* R1  = ow + E;                   // [N*256] feat_in|feat_out -> f -> k
  float* feat_in  = R1;
  float* feat_out = R1 + (long)N * D;
  float* a2   = R1 + (long)N * 2 * D;    // [N*256] a2 -> g2
  float* fbuf = R1;
  float* g2buf = a2;
  float* feat = a2 + (long)N * 2 * D;    // [N*128]
  float* kbuf = R1;                      // reuse after f consumed
  float* qry   = feat + (long)N * D;     // [B*128]
  float* elast = qry + (long)B * D;      // [B*128]

  edge_norm_kernel<<<B, 128, 0, stream>>>(src, dst, w, iw, ow);

  gemm_wmma_kernel<128><<<N / 128, 256, 0, stream>>>(h_i, W_in,  b_in,  feat_in,  128, 0);
  gemm_wmma_kernel<128><<<N / 128, 256, 0, stream>>>(h_i, W_out, b_out, feat_out, 128, 0);

  aggregate_kernel<<<B, 128, 0, stream>>>(src, dst, iw, ow, feat_in, feat_out, a2);

  gemm_wmma_kernel<256><<<N / 128, 256, 0, stream>>>(a2,  Wg_in,  bg_in,   fbuf,  256, 0);
  gemm_wmma_kernel<256><<<N / 128, 256, 0, stream>>>(h_i, Wg_out, nullptr, g2buf, 128, 0);

  gate_kernel<<<(N * 128) / 256, 256, 0, stream>>>(fbuf, g2buf, h_i, a_gru, feat, (long)N * 128);

  gemm_wmma_kernel<128><<<N / 128, 256, 0, stream>>>(feat, W_key, nullptr, kbuf, 128, 0);

  gather_last_kernel<<<(B * 128) / 256, 256, 0, stream>>>(feat, last_nodes, elast, out);

  gemm_wmma_kernel<128><<<B / 128, 256, 0, stream>>>(HKGU,  W_user, b_user, qry, 128, 0);
  gemm_wmma_kernel<128><<<B / 128, 256, 0, stream>>>(elast, W_last, nullptr, qry, 128, 1);

  attention_kernel<<<B, 128, 0, stream>>>(qry, kbuf, feat, W_e, cnt, a_att, out);
}
